// BlockSONAR_24189255811085
// MI455X (gfx1250) — matrix operand order repaired and checked by reading, not compiled, not run on hardware
//
#include <hip/hip_runtime.h>

#define EPS 0.01f
#define LDT 72   // padded row stride (bf16) for weight tiles in LDS
#define LDX 64   // unpadded row stride (bf16) for x tiles (async-DMA'd contiguously)

#if defined(__gfx1250__) && __has_builtin(__builtin_amdgcn_global_load_async_to_lds_b128)
#define HAVE_ASYNC_LDS 1
#else
#define HAVE_ASYNC_LDS 0
#endif

typedef __attribute__((ext_vector_type(16))) __bf16 v16bf;
typedef __attribute__((ext_vector_type(8)))  __bf16 v8bf;
typedef __attribute__((ext_vector_type(8)))  float  v8f;

#if HAVE_ASYNC_LDS
typedef int v4i_vs __attribute__((vector_size(16)));
typedef __attribute__((address_space(1))) v4i_vs* g_v4i_ptr;
typedef __attribute__((address_space(3))) v4i_vs* l_v4i_ptr;
#endif

__device__ __forceinline__ void wait_async0() {
#if HAVE_ASYNC_LDS
#if __has_builtin(__builtin_amdgcn_s_wait_asynccnt)
  __builtin_amdgcn_s_wait_asynccnt(0);
#else
  asm volatile("s_wait_asynccnt 0x0" ::: "memory");
#endif
#endif
}

// async-copy (or fallback-copy) one 16KB x-tile (128 rows x 64 bf16, contiguous) into LDS
__device__ __forceinline__ void load_xtile(__bf16* xt, const __bf16* X, int row0, int tid) {
#if HAVE_ASYNC_LDS
  char* src = (char*)(void*)(X + (size_t)row0 * 64);
  char* dst = (char*)xt;
#pragma unroll
  for (int i = 0; i < 4; ++i) {
    int c = tid + i * 256;
    __builtin_amdgcn_global_load_async_to_lds_b128(
        (g_v4i_ptr)(src + (size_t)c * 16),
        (l_v4i_ptr)(dst + c * 16), 0, 0);
  }
  wait_async0();
#else
  const ulonglong2* src = (const ulonglong2*)(X + (size_t)row0 * 64);
  ulonglong2* dst = (ulonglong2*)xt;
  for (int c = tid; c < 1024; c += 256) dst[c] = src[c];
#endif
}

// ---- A fragment: 16x32 bf16, lane holds row M, K = 8*half+{0..7} and 16+8*half+{0..7}
__device__ __forceinline__ v16bf load_afrag(const __bf16* row, int hf, int kblock) {
  const int k0 = kblock * 32 + hf * 8;
  v8bf lo = *(const v8bf*)(row + k0);
  v8bf hi = *(const v8bf*)(row + k0 + 16);
  v16bf f;
#pragma unroll
  for (int i = 0; i < 8; ++i) { f[i] = lo[i]; f[i + 8] = hi[i]; }
  return f;
}

// ---- B fragment: 32x16 bf16, lane holds column N = ncol0 + lane%16, K = 8*half+{0..7} and 16+8*half+{0..7}
__device__ __forceinline__ v16bf load_bfrag(const __bf16* wlds, int lane, int kblock, int ncol0) {
  const __bf16* p = wlds + (size_t)(kblock * 32 + (lane >> 4) * 8) * LDT + ncol0 + (lane & 15);
  v16bf f;
#pragma unroll
  for (int i = 0; i < 8; ++i) { f[i] = p[i * LDT]; f[i + 8] = p[(i + 16) * LDT]; }
  return f;
}

__device__ __forceinline__ float gelu_tanh(float x) {
  float x3 = x * x * x;
  return 0.5f * x * (1.0f + tanhf(0.7978845608f * (x + 0.044715f * x3)));
}

// ============ generic GEMM: out = act(Xbf @ W + b) (+resid) [+ bf16 mirror] ============
template <int ACT, bool HAS_RESID, bool OUT_BF>
__global__ __launch_bounds__(256) void gemm64_kernel(
    const __bf16* __restrict__ X, const float* __restrict__ W,
    const float* __restrict__ bias, const float* __restrict__ resid,
    float* __restrict__ out, __bf16* __restrict__ out_bf,
    int n_rows, int out_w) {
  __shared__ __align__(16) __bf16 xt[128 * LDX];
  __shared__ __align__(16) __bf16 wt[64 * LDT];
  const int tid = threadIdx.x;
  const int row0 = blockIdx.x * 128;

  load_xtile(xt, X, row0, tid);
  for (int idx = tid; idx < 64 * out_w; idx += 256) {
    int k = idx / out_w, n = idx % out_w;
    wt[k * LDT + n] = (__bf16)W[k * out_w + n];
  }
  __syncthreads();

  const int lane = tid & 31, wave = tid >> 5;
  const int hf = lane >> 4, lq = lane & 15;
  const __bf16* arow = &xt[(wave * 16 + lq) * LDX];
  v16bf a0 = load_afrag(arow, hf, 0);
  v16bf a1 = load_afrag(arow, hf, 1);
  const int ntiles = out_w >> 4;

  v8f acc[4];
  for (int nt = 0; nt < ntiles; ++nt) {
    v16bf b0 = load_bfrag(wt, lane, 0, nt * 16);
    v16bf b1 = load_bfrag(wt, lane, 1, nt * 16);
    v8f c = {};
    c = __builtin_amdgcn_wmma_f32_16x16x32_bf16(false, a0, false, b0, (short)0, c, false, false);
    c = __builtin_amdgcn_wmma_f32_16x16x32_bf16(false, a1, false, b1, (short)0, c, false, false);
    acc[nt] = c;
  }

  const int rbase = row0 + wave * 16 + hf * 8;
  if (row0 + 128 <= n_rows) {           // uniform: full tile, no per-lane guards
    for (int nt = 0; nt < ntiles; ++nt) {
      float bv = bias[nt * 16 + lq];
#pragma unroll
      for (int j = 0; j < 8; ++j) {
        float o = acc[nt][j] + bv;
        if (ACT == 1) o = fmaxf(o, 0.0f);
        if (ACT == 2) o = gelu_tanh(o);
        size_t oi = (size_t)(rbase + j) * out_w + nt * 16 + lq;
        if (HAS_RESID) o += resid[oi];
        out[oi] = o;
        if (OUT_BF) out_bf[oi] = (__bf16)o;
      }
    }
  } else {                               // tail tile
    for (int nt = 0; nt < ntiles; ++nt) {
      float bv = bias[nt * 16 + lq];
#pragma unroll
      for (int j = 0; j < 8; ++j) {
        int gr = rbase + j;
        if (gr < n_rows) {
          float o = acc[nt][j] + bv;
          if (ACT == 1) o = fmaxf(o, 0.0f);
          if (ACT == 2) o = gelu_tanh(o);
          size_t oi = (size_t)gr * out_w + nt * 16 + lq;
          if (HAS_RESID) o += resid[oi];
          out[oi] = o;
          if (OUT_BF) out_bf[oi] = (__bf16)o;
        }
      }
    }
  }
}

// ============ fused SONAR node pass: lx, relu(diss), forc, a, b; zero agg/deg ============
__global__ __launch_bounds__(256) void sonar_pre_kernel(
    const __bf16* __restrict__ X,
    const float* __restrict__ lin_w, const float* __restrict__ lin_b,
    const float* __restrict__ diss_w, const float* __restrict__ diss_b,
    const float* __restrict__ forc_w, const float* __restrict__ forc_b,
    const float* __restrict__ res_w,  // wa = res_w[0..63], wb = res_w[64..127]
    float* __restrict__ lx, float* __restrict__ diss, float* __restrict__ forc,
    float* __restrict__ avec, float* __restrict__ bvec,
    float* __restrict__ agg, float* __restrict__ deg, int n_rows) {
  __shared__ __align__(16) __bf16 xt[128 * LDX];
  __shared__ __align__(16) __bf16 wt[3 * 64 * LDT];
  __shared__ float wab[128];
  const int tid = threadIdx.x;
  const int row0 = blockIdx.x * 128;

  load_xtile(xt, X, row0, tid);
  for (int idx = tid; idx < 3 * 64 * 64; idx += 256) {
    int mat = idx >> 12, rem = idx & 4095;
    int k = rem >> 6, n = rem & 63;
    const float* Wm = (mat == 0) ? lin_w : ((mat == 1) ? diss_w : forc_w);
    wt[(size_t)mat * 64 * LDT + k * LDT + n] = (__bf16)Wm[k * 64 + n];
  }
  if (tid < 128) wab[tid] = res_w[tid];
  // zero agg/deg for this tile's nodes
  for (int idx = tid; idx < 128 * 64; idx += 256) {
    int gr = row0 + (idx >> 6);
    if (gr < n_rows) agg[(size_t)gr * 64 + (idx & 63)] = 0.0f;
  }
  if (tid < 128) {
    int gr = row0 + tid;
    if (gr < n_rows) deg[gr] = 0.0f;
  }
  __syncthreads();

  const int lane = tid & 31, wave = tid >> 5;
  const int hf = lane >> 4, lq = lane & 15;
  const __bf16* arow = &xt[(wave * 16 + lq) * LDX];
  v16bf a0 = load_afrag(arow, hf, 0);
  v16bf a1 = load_afrag(arow, hf, 1);
  const int rbase = row0 + wave * 16 + hf * 8;
  const bool full = (row0 + 128 <= n_rows);

  for (int mat = 0; mat < 3; ++mat) {
    const __bf16* wm = &wt[(size_t)mat * 64 * LDT];
    const float* bm = (mat == 0) ? lin_b : ((mat == 1) ? diss_b : forc_b);
    float* om = (mat == 0) ? lx : ((mat == 1) ? diss : forc);
    v8f acc[4];
    for (int nt = 0; nt < 4; ++nt) {
      v16bf b0 = load_bfrag(wm, lane, 0, nt * 16);
      v16bf b1 = load_bfrag(wm, lane, 1, nt * 16);
      v8f c = {};
      c = __builtin_amdgcn_wmma_f32_16x16x32_bf16(false, a0, false, b0, (short)0, c, false, false);
      c = __builtin_amdgcn_wmma_f32_16x16x32_bf16(false, a1, false, b1, (short)0, c, false, false);
      acc[nt] = c;
    }
    if (full) {
      for (int nt = 0; nt < 4; ++nt) {
        float bv = bm[nt * 16 + lq];
#pragma unroll
        for (int j = 0; j < 8; ++j) {
          float o = acc[nt][j] + bv;
          if (mat == 1) o = fmaxf(o, 0.0f);
          om[(size_t)(rbase + j) * 64 + nt * 16 + lq] = o;
        }
      }
    } else {
      for (int nt = 0; nt < 4; ++nt) {
        float bv = bm[nt * 16 + lq];
#pragma unroll
        for (int j = 0; j < 8; ++j) {
          int gr = rbase + j;
          if (gr < n_rows) {
            float o = acc[nt][j] + bv;
            if (mat == 1) o = fmaxf(o, 0.0f);
            om[(size_t)gr * 64 + nt * 16 + lq] = o;
          }
        }
      }
    }
  }

  // a = x@wa, b = x@wb
  if (tid < 128) {
    int gr = row0 + tid;
    if (gr < n_rows) {
      float sa = 0.0f, sb = 0.0f;
      const __bf16* xr = &xt[tid * LDX];
#pragma unroll
      for (int k = 0; k < 64; ++k) {
        float xv = (float)xr[k];
        sa += xv * wab[k];
        sb += xv * wab[64 + k];
      }
      avec[gr] = sa;
      bvec[gr] = sb;
    }
  }
}

// ============ fp32 -> bf16 (zero-fill the padded tail) ============
__global__ __launch_bounds__(256) void tobf16_kernel(
    const float* __restrict__ src, __bf16* __restrict__ dst, long long n, long long npad) {
  long long i = (long long)blockIdx.x * 256 + threadIdx.x;
  if (i >= npad) return;
  dst[i] = (__bf16)(i < n ? src[i] : 0.0f);
}

// ============ edge pass 1: r = relu(a[row]+b[col]+rb)*(row!=col); deg[row] += r ============
__global__ __launch_bounds__(256) void edge_r_kernel(
    const int* __restrict__ erow, const int* __restrict__ ecol,
    const float* __restrict__ avec, const float* __restrict__ bvec,
    const float* __restrict__ resb, float* __restrict__ r,
    float* __restrict__ deg, int E) {
  int e = blockIdx.x * 256 + threadIdx.x;
  if (e >= E) return;
  int ri = erow[e], ci = ecol[e];
  float rv = avec[ri] + bvec[ci] + resb[0];
  rv = fmaxf(rv, 0.0f);
  if (ri == ci) rv = 0.0f;
  r[e] = rv;
  if (rv != 0.0f) atomicAdd(&deg[ri], rv);
}

__global__ __launch_bounds__(256) void dinv_kernel(
    const float* __restrict__ deg, float* __restrict__ dinv, int n) {
  int i = blockIdx.x * 256 + threadIdx.x;
  if (i >= n) return;
  float d = deg[i];
  dinv[i] = (d > 0.0f) ? rsqrtf(d) : 0.0f;
}

// ============ edge pass 2: agg[col] += wn * lx[row]  (16 lanes x float4 per edge) ============
__global__ __launch_bounds__(256) void edge_scatter_kernel(
    const int* __restrict__ erow, const int* __restrict__ ecol,
    const float* __restrict__ r, const float* __restrict__ dinv,
    const float* __restrict__ lx, float* __restrict__ agg, int E) {
  long long t = (long long)blockIdx.x * 256 + threadIdx.x;
  int e = (int)(t >> 4);
  int q = (int)(t & 15);
  if (e >= E) return;
  int ri = erow[e], ci = ecol[e];
  __builtin_prefetch(lx + (size_t)ri * 64, 0, 1);
  float wn = dinv[ri] * r[e] * dinv[ci];
  if (wn == 0.0f) return;
  float4 v = ((const float4*)(lx + (size_t)ri * 64))[q];
  float* dst = agg + (size_t)ci * 64 + q * 4;
  atomicAdd(dst + 0, wn * v.x);
  atomicAdd(dst + 1, wn * v.y);
  atomicAdd(dst + 2, wn * v.z);
  atomicAdd(dst + 3, wn * v.w);
}

// ===== node update: v -= EPS*((lx-agg)+diss*v-forc); h += EPS*v; keep bf16 mirror of h =====
__global__ __launch_bounds__(256) void update_kernel(
    float* __restrict__ h, float* __restrict__ v,
    const float* __restrict__ lx, const float* __restrict__ agg,
    const float* __restrict__ diss, const float* __restrict__ forc,
    __bf16* __restrict__ h_bf, int n) {
  long long t = (long long)blockIdx.x * 256 + threadIdx.x;
  if (t >= (long long)n * 16) return;
  float4 lxv = ((const float4*)lx)[t];
  float4 ag  = ((const float4*)agg)[t];
  float4 dv  = ((const float4*)diss)[t];
  float4 fv  = ((const float4*)forc)[t];
  float4 vv  = ((const float4*)v)[t];
  float4 hv  = ((const float4*)h)[t];
  vv.x -= EPS * ((lxv.x - ag.x) + dv.x * vv.x - fv.x); hv.x += EPS * vv.x;
  vv.y -= EPS * ((lxv.y - ag.y) + dv.y * vv.y - fv.y); hv.y += EPS * vv.y;
  vv.z -= EPS * ((lxv.z - ag.z) + dv.z * vv.z - fv.z); hv.z += EPS * vv.z;
  vv.w -= EPS * ((lxv.w - ag.w) + dv.w * vv.w - fv.w); hv.w += EPS * vv.w;
  ((float4*)v)[t] = vv;
  ((float4*)h)[t] = hv;
  __bf16* hb = h_bf + t * 4;
  hb[0] = (__bf16)hv.x; hb[1] = (__bf16)hv.y; hb[2] = (__bf16)hv.z; hb[3] = (__bf16)hv.w;
}

extern "C" void kernel_launch(void* const* d_in, const int* in_sizes, int n_in,
                              void* d_out, int out_size, void* d_ws, size_t ws_size,
                              hipStream_t stream) {
  const float* x_in   = (const float*)d_in[0];
  const int*   edge   = (const int*)d_in[1];
  const float* enc_w  = (const float*)d_in[2];
  const float* enc_b  = (const float*)d_in[3];
  const float* vel_w  = (const float*)d_in[4];
  const float* vel_b  = (const float*)d_in[5];
  const float* res_w  = (const float*)d_in[6];
  const float* res_b  = (const float*)d_in[7];
  const float* lin_w  = (const float*)d_in[8];
  const float* lin_b  = (const float*)d_in[9];
  const float* diss_w = (const float*)d_in[10];
  const float* diss_b = (const float*)d_in[11];
  const float* forc_w = (const float*)d_in[12];
  const float* forc_b = (const float*)d_in[13];
  const float* mlp_w1 = (const float*)d_in[14];
  const float* mlp_b1 = (const float*)d_in[15];
  const float* mlp_w2 = (const float*)d_in[16];
  const float* mlp_b2 = (const float*)d_in[17];
  const float* dec_w  = (const float*)d_in[18];
  const float* dec_b  = (const float*)d_in[19];

  const int N = in_sizes[0] / 64;
  const int E = in_sizes[1] / 2;
  const int* erow = edge;
  const int* ecol = edge + E;

  float* ws = (float*)d_ws;
  size_t N64 = (size_t)N * 64;
  int gemm_grid = (N + 127) / 128;
  size_t NPAD64 = (size_t)gemm_grid * 128 * 64;  // mirror length incl. tile padding

  float* h    = ws;
  float* v    = h + N64;
  float* lx   = v + N64;       // also MLP intermediate t
  float* dsb  = lx + N64;
  float* fob  = dsb + N64;
  float* agg  = fob + N64;
  float* avec = agg + N64;
  float* bvec = avec + N;
  float* deg  = bvec + N;
  float* dnv  = deg + N;
  float* rbuf = dnv + N;
  __bf16* xbf = (__bf16*)(rbuf + E);
  __bf16* hbf = xbf + NPAD64;
  __bf16* tbf = hbf + NPAD64;

  dim3 blk(256);
  int node4     = (int)((N64 / 4 + 255) / 256);
  int edge_grid = (E + 255) / 256;
  int sc_grid   = (int)(((long long)E * 16 + 255) / 256);
  int n_grid    = (N + 255) / 256;
  int cvt_grid  = (int)((NPAD64 + 255) / 256);

  // bf16 mirror of the input features
  tobf16_kernel<<<cvt_grid, blk, 0, stream>>>(x_in, xbf, (long long)N64, (long long)NPAD64);

  // encoder: h = x @ enc_w + enc_b (+ bf16 mirror)
  gemm64_kernel<0, false, true><<<gemm_grid, blk, 0, stream>>>(
      xbf, enc_w, enc_b, nullptr, h, hbf, N, 64);

  for (int j = 0; j < 2; ++j) {
    // v = h @ vel_w + vel_b
    gemm64_kernel<0, false, false><<<gemm_grid, blk, 0, stream>>>(
        hbf, vel_w + j * 4096, vel_b + j * 64, nullptr, v, nullptr, N, 64);
    for (int it = 0; it < 4; ++it) {
      sonar_pre_kernel<<<gemm_grid, blk, 0, stream>>>(
          hbf, lin_w + j * 4096, lin_b + j * 64, diss_w + j * 4096, diss_b + j * 64,
          forc_w + j * 4096, forc_b + j * 64, res_w + j * 128,
          lx, dsb, fob, avec, bvec, agg, deg, N);
      edge_r_kernel<<<edge_grid, blk, 0, stream>>>(erow, ecol, avec, bvec, res_b + j,
                                                   rbuf, deg, E);
      dinv_kernel<<<n_grid, blk, 0, stream>>>(deg, dnv, N);
      edge_scatter_kernel<<<sc_grid, blk, 0, stream>>>(erow, ecol, rbuf, dnv, lx, agg, E);
      update_kernel<<<node4, blk, 0, stream>>>(h, v, lx, agg, dsb, fob, hbf, N);
    }
    // MLP with residual: t = gelu(h@w1+b1); h = h + t@w2 + b2
    gemm64_kernel<2, false, true><<<gemm_grid, blk, 0, stream>>>(
        hbf, mlp_w1 + j * 4096, mlp_b1 + j * 64, nullptr, lx, tbf, N, 64);
    gemm64_kernel<0, true, true><<<gemm_grid, blk, 0, stream>>>(
        tbf, mlp_w2 + j * 4096, mlp_b2 + j * 64, h, h, hbf, N, 64);
  }
  // decoder
  gemm64_kernel<0, false, false><<<gemm_grid, blk, 0, stream>>>(
      hbf, dec_w, dec_b, nullptr, (float*)d_out, nullptr, N, 32);
}
